// Encoder_27986006901356
// MI455X (gfx1250) — matrix operand order repaired and checked by reading, not compiled, not run on hardware
//
#include <hip/hip_runtime.h>
#include <hip/hip_bf16.h>
#include <stdint.h>

// ---------------------------------------------------------------------------
// 2-layer LSTM (B=32, T=512, F=H=1024) for gfx1250 (MI455X).
//   Phase A: xz = x@W + b   (bf16 WMMA GEMM, f32 accumulate, bias-init)
//   Phase B: persistent recurrence kernel, 32 WGs, grid barrier per step,
//            ping-pong h buffers, c kept in fp32 registers.
// U pre-packed into the wave32 WMMA B-matrix lane layout (2x global_load_b128
// per 32x16 K-tile straight from L2). h staged to LDS each step with
// GLOBAL_LOAD_ASYNC_TO_LDS_B128 (ASYNCcnt). Gate nonlinearities use
// v_exp_f32 + v_rcp_f32 (no IEEE divide expansion).
// ---------------------------------------------------------------------------

typedef __attribute__((ext_vector_type(16))) __bf16 v16bf;
typedef __attribute__((ext_vector_type(8)))  float  v8f;

static constexpr int BB = 32, TT = 512, HH = 1024, FOURH = 4096;
static constexpr int BT = BB * TT;

union V16U { uint4 u[2]; v16bf v; };
union V8U  { v8f v; float f[8]; };

__device__ __forceinline__ uint32_t f2bf1(float f) {
  uint32_t u = __builtin_bit_cast(uint32_t, f);
  return (u + 0x7FFFu + ((u >> 16) & 1u)) >> 16;   // RNE bf16
}
__device__ __forceinline__ uint32_t f2bf2(float lo, float hi) {
  return f2bf1(lo) | (f2bf1(hi) << 16);
}
// Fast gate nonlinearities: 1 trans (exp) + 1 trans (rcp) + fma, no v_div_*.
__device__ __forceinline__ float sigmoidf_(float x) {
  return __builtin_amdgcn_rcpf(1.0f + __expf(-x));
}
__device__ __forceinline__ float tanhf_(float x) {
  // tanh(x) = 2*sigmoid(2x) - 1
  return __builtin_fmaf(2.0f, __builtin_amdgcn_rcpf(1.0f + __expf(-2.0f * x)), -1.0f);
}

// LDS byte offset of a shared-memory pointer (LDS aperture: addr[31:0]).
__device__ __forceinline__ uint32_t lds_off32(const void* p) {
  return (uint32_t)(uintptr_t)p;
}
// Async 16B copy: global -> LDS, tracked by ASYNCcnt (no VGPR data path).
__device__ __forceinline__ void async_copy_b128(uint32_t lds_byte_off,
                                                const void* gptr) {
  uint64_t ga = (uint64_t)(uintptr_t)gptr;
  asm volatile("global_load_async_to_lds_b128 %0, %1, off"
               :: "v"(lds_byte_off), "v"(ga)
               : "memory");
}
__device__ __forceinline__ void async_wait0() {
  asm volatile("s_wait_asynccnt 0x0" ::: "memory");
}

// -------------------------------- utils ------------------------------------
__global__ void zero_kernel(uint32_t* p, int n) {
  int i = blockIdx.x * 256 + threadIdx.x;
  if (i < n) p[i] = 0u;
}

// Pack U [1024 x 4096] fp32 row-major -> bf16 WMMA-B tiles.
// tile tau = kt*256 + nt (kt: 32 K-tiles, nt: 256 16-col N-tiles).
// Within tile, dword index = lane*8 + v:
//   lane<16 : col = nt*16+lane,    k = kt*32 + 2v      (K pairs 0..15)
//   lane>=16: col = nt*16+lane-16, k = kt*32 + 16 + 2v (K pairs 16..31)
__global__ void pack_weights(const float* __restrict__ U, uint32_t* __restrict__ Up) {
  int idx = blockIdx.x * 256 + threadIdx.x;      // 32*256*32*8 = 2,097,152
  int v    = idx & 7;
  int lane = (idx >> 3) & 31;
  int tile = idx >> 8;
  int nt   = tile & 255;
  int kt   = tile >> 8;
  int col  = nt * 16 + (lane & 15);
  int k0   = kt * 32 + ((lane & 16) ? 16 : 0) + 2 * v;
  Up[idx] = f2bf2(U[(size_t)k0 * FOURH + col], U[(size_t)(k0 + 1) * FOURH + col]);
}

// ------------------------------ phase A GEMM --------------------------------
// out[16384 x 4096] = A[16384 x 1024] @ W[1024 x 4096] + bias
// WG tile 64x128, 8 waves, each wave: 1 M-tile x 4 N-tiles.
template <bool ABF16>
__global__ __launch_bounds__(256) void gemm_bias(const void* __restrict__ Ain,
                                                 const float* __restrict__ W,
                                                 const float* __restrict__ bias,
                                                 float* __restrict__ out) {
  __shared__ alignas(16) uint16_t alds[64 * 32];
  __shared__ alignas(16) uint32_t blds[2048];     // 8 packed 32x16 B tiles

  const int tid = threadIdx.x;
  const int rowbase = blockIdx.x * 64;
  const int colbase = blockIdx.y * 128;
  const int w = tid >> 5, lane = tid & 31;
  const int mt = w & 3, nh = w >> 2;
  const int lm = lane & 15;
  const bool hiLane = (lane & 16) != 0;

  V8U acc[4];
#pragma unroll
  for (int j = 0; j < 4; j++) {
    float bv = bias[colbase + nh * 64 + j * 16 + lm];
#pragma unroll
    for (int r = 0; r < 8; r++) acc[j].f[r] = bv;
  }

  for (int kt = 0; kt < 32; kt++) {
    const int kb = kt * 32;
    // stage A (64x32)
    {
      int r = tid >> 2, kg = tid & 3;
      if (!ABF16) {
        const float* A = (const float*)Ain;
        const float4* src = (const float4*)(A + (size_t)(rowbase + r) * HH + kb + kg * 8);
        float4 a0 = src[0], a1 = src[1];
        uint4 d;
        d.x = f2bf2(a0.x, a0.y); d.y = f2bf2(a0.z, a0.w);
        d.z = f2bf2(a1.x, a1.y); d.w = f2bf2(a1.z, a1.w);
        *(uint4*)&alds[r * 32 + kg * 8] = d;
      } else {
        // bf16 input: async DMA straight into LDS, no VGPR round-trip
        const uint16_t* A = (const uint16_t*)Ain;
        async_copy_b128(lds_off32(&alds[r * 32 + kg * 8]),
                        A + (size_t)(rowbase + r) * HH + kb + kg * 8);
      }
    }
    // stage B into packed tile layout (8 n-groups of 16 cols)
#pragma unroll
    for (int v = 0; v < 8; v++) {
      int ng = tid >> 5, ln = tid & 31;
      int col = colbase + ng * 16 + (ln & 15);
      int k = kb + ((ln & 16) ? 16 : 0) + 2 * v;
      blds[(ng * 32 + ln) * 8 + v] =
          f2bf2(W[(size_t)k * FOURH + col], W[(size_t)(k + 1) * FOURH + col]);
    }
    if (ABF16) async_wait0();
    __syncthreads();

    V16U av;
    {
      int m = mt * 16 + lm;
      int k0 = hiLane ? 16 : 0, k1 = hiLane ? 24 : 8;
      av.u[0] = *(const uint4*)&alds[m * 32 + k0];
      av.u[1] = *(const uint4*)&alds[m * 32 + k1];
    }
#pragma unroll
    for (int j = 0; j < 4; j++) {
      int nt = nh * 4 + j;
      V16U bvv;
      bvv.u[0] = *(const uint4*)&blds[nt * 256 + lane * 8];
      bvv.u[1] = *(const uint4*)&blds[nt * 256 + lane * 8 + 4];
      acc[j].v = __builtin_amdgcn_wmma_f32_16x16x32_bf16(
          false, av.v, false, bvv.v, (short)0, acc[j].v, false, false);
    }
    __syncthreads();
  }

#pragma unroll
  for (int j = 0; j < 4; j++) {
    int col = colbase + nh * 64 + j * 16 + lm;
#pragma unroll
    for (int r = 0; r < 8; r++) {
      int row = rowbase + mt * 16 + r + (hiLane ? 8 : 0);
      out[(size_t)row * FOURH + col] = acc[j].f[r];
    }
  }
}

// ---------------------------- phase B recurrence ----------------------------
// 32 WGs x 128 threads, persistent over t. WG wg owns h columns [wg*32, wg*32+32).
// Wave (mhalf, nsub): 16 batch rows x 16 h-cols, all 4 gates.
__global__ __launch_bounds__(128) void lstm_recur(const float* __restrict__ xz,
                                                  const uint32_t* __restrict__ Up,
                                                  uint16_t* __restrict__ hA,
                                                  uint16_t* __restrict__ hB,
                                                  uint16_t* __restrict__ seq_out,
                                                  float* __restrict__ hc_out,
                                                  int* __restrict__ bar) {
  __shared__ alignas(16) uint16_t hlds[32 * 1024];   // full h, bf16, 64 KB

  const int tid = threadIdx.x;
  const int wg = blockIdx.x;
  const int w = tid >> 5, lane = tid & 31;
  const int mhalf = w & 1, nsub = w >> 1;
  const int lm = lane & 15;
  const bool hiLane = (lane & 16) != 0;
  const int ncol = wg * 32 + nsub * 16 + lm;     // h column
  const int ntbase = wg * 2 + nsub;              // 16-col tile within a gate block
  const uint32_t lbase = lds_off32(&hlds[0]);

  V8U creg;
#pragma unroll
  for (int r = 0; r < 8; r++) creg.f[r] = 0.0f;

  for (int t = 0; t < TT; t++) {
    const uint16_t* hsrc = (t & 1) ? hB : hA;    // t=0 reads zeroed hA
    uint16_t* hdst = (t & 1) ? hA : hB;

    // stage h_t into LDS via async DMA (64 KB, 16B per lane per issue)
#pragma unroll 4
    for (int i = 0; i < 32; i++) {
      int e = i * 128 + tid;
      async_copy_b128(lbase + (uint32_t)e * 16, (const char*)hsrc + (size_t)e * 16);
    }
    async_wait0();
    __syncthreads();

    // init gate accumulators from precomputed xz_t (fp32), static indexing
    V8U acc[4];
#pragma unroll
    for (int g = 0; g < 4; g++) {
      int zc = g * HH + ncol;
#pragma unroll
      for (int r = 0; r < 8; r++) {
        int brow = mhalf * 16 + r + (hiLane ? 8 : 0);
        acc[g].f[r] = xz[((size_t)brow * TT + t) * FOURH + zc];
      }
    }

    // z += h @ U : K loop, A from LDS, B (pre-packed) from L2
    for (int kt = 0; kt < 32; kt++) {
      V16U av;
      int m = mhalf * 16 + lm;
      int k0 = kt * 32 + (hiLane ? 16 : 0);
      int k1 = kt * 32 + (hiLane ? 24 : 8);
      av.u[0] = *(const uint4*)&hlds[m * 1024 + k0];
      av.u[1] = *(const uint4*)&hlds[m * 1024 + k1];
#pragma unroll
      for (int g = 0; g < 4; g++) {
        int tile = kt * 256 + g * 64 + ntbase;
        const uint4* bp = (const uint4*)(Up + (size_t)tile * 256 + lane * 8);
        V16U bvv; bvv.u[0] = bp[0]; bvv.u[1] = bp[1];
        acc[g].v = __builtin_amdgcn_wmma_f32_16x16x32_bf16(
            false, av.v, false, bvv.v, (short)0, acc[g].v, false, false);
      }
    }

    // gates + state update (c stays fp32 in registers), static indexing
#pragma unroll
    for (int r = 0; r < 8; r++) {
      int brow = mhalf * 16 + r + (hiLane ? 8 : 0);
      float ig = sigmoidf_(acc[0].f[r]);
      float fg = sigmoidf_(acc[1].f[r]);
      float gg = tanhf_(acc[2].f[r]);
      float og = sigmoidf_(acc[3].f[r]);
      float c = fg * creg.f[r] + ig * gg;
      creg.f[r] = c;
      float hf = og * tanhf_(c);
      hdst[(size_t)brow * HH + ncol] = (uint16_t)f2bf1(hf);
      if (seq_out) seq_out[((size_t)brow * TT + t) * HH + ncol] = (uint16_t)f2bf1(hf);
      if (hc_out && t == TT - 1) {
        hc_out[(size_t)brow * HH + ncol] = hf;                 // h1
        hc_out[32768 + (size_t)brow * HH + ncol] = hf;         // h1 (dup)
        hc_out[65536 + (size_t)brow * HH + ncol] = c;          // c1
      }
    }

    // grid-wide barrier: everyone's h_{t+1} visible before next stage
    __syncthreads();
    if (tid == 0) {
      __threadfence();
      atomicAdd(&bar[t], 1);
      volatile int* p = bar + t;
      while (*p < (int)gridDim.x) { __builtin_amdgcn_s_sleep(1); }
      __threadfence();
    }
    __syncthreads();
  }
}

// --------------------------------- launcher ---------------------------------
extern "C" void kernel_launch(void* const* d_in, const int* in_sizes, int n_in,
                              void* d_out, int out_size, void* d_ws, size_t ws_size,
                              hipStream_t stream) {
  (void)in_sizes; (void)n_in; (void)out_size; (void)ws_size;
  const float* x  = (const float*)d_in[0];
  const float* W0 = (const float*)d_in[1];
  const float* U0 = (const float*)d_in[2];
  const float* b0 = (const float*)d_in[3];
  const float* W1 = (const float*)d_in[4];
  const float* U1 = (const float*)d_in[5];
  const float* b1 = (const float*)d_in[6];
  float* out = (float*)d_out;

  char* ws = (char*)d_ws;
  size_t off = 0;
  float*    xz   = (float*)(ws + off);    off += (size_t)BT * FOURH * 4;   // 256 MB
  uint32_t* U0p  = (uint32_t*)(ws + off); off += (size_t)HH * FOURH * 2;   // 8 MB
  uint32_t* U1p  = (uint32_t*)(ws + off); off += (size_t)HH * FOURH * 2;   // 8 MB
  uint16_t* seq0 = (uint16_t*)(ws + off); off += (size_t)BT * HH * 2;      // 32 MB
  uint16_t* hA   = (uint16_t*)(ws + off); off += 65536;
  uint16_t* hB   = (uint16_t*)(ws + off); off += 65536;
  int*      bar0 = (int*)(ws + off);      off += 2048;
  int*      bar1 = (int*)(ws + off);      off += 2048;

  // zero hA, hB, bar0, bar1 (contiguous region, 33792 dwords)
  zero_kernel<<<132, 256, 0, stream>>>((uint32_t*)hA, 33792);

  pack_weights<<<8192, 256, 0, stream>>>(U0, U0p);
  pack_weights<<<8192, 256, 0, stream>>>(U1, U1p);

  dim3 ggrid(BT / 64, FOURH / 128);   // (256, 32)
  gemm_bias<false><<<ggrid, 256, 0, stream>>>(x, W0, b0, xz);
  lstm_recur<<<32, 128, 0, stream>>>(xz, U0p, hA, hB, seq0, nullptr, bar0);

  // re-zero hA (h0 = 0 for layer 1); hB is fully overwritten at t=0
  zero_kernel<<<64, 256, 0, stream>>>((uint32_t*)hA, 16384);

  gemm_bias<true><<<ggrid, 256, 0, stream>>>(seq0, W1, b1, xz);
  lstm_recur<<<32, 128, 0, stream>>>(xz, U1p, hA, hB, nullptr, out, bar1);
}
